// all_atom_view_graph_54030688584318
// MI455X (gfx1250) — matrix operand order repaired and checked by reading, not compiled, not run on hardware
//
#include <hip/hip_runtime.h>

typedef __attribute__((ext_vector_type(16))) __bf16 v16bf;
typedef __attribute__((ext_vector_type(8)))  float  v8f;

#define HDIM   128
#define G_     16
#define NCPLX  2048
#define NNODES (G_ * NCPLX)      // 32768
#define NEDGE  (G_ * 16384)      // 262144

// ---------------- scalar helpers ----------------
__device__ __forceinline__ __bf16 f2bf(float f) {
  union { float f; unsigned u; } v; v.f = f;
  unsigned r = (v.u + 0x7FFFu + ((v.u >> 16) & 1u)) >> 16;
  union { unsigned short s; __bf16 b; } o; o.s = (unsigned short)r;
  return o.b;
}
__device__ __forceinline__ float bf2f(__bf16 b) {
  union { __bf16 b; unsigned short s; } i; i.b = b;
  union { unsigned u; float f; } o; o.u = ((unsigned)i.s) << 16;
  return o.f;
}
__device__ __forceinline__ float silu_f(float x) {
  return x * (1.0f / (1.0f + __expf(-x)));
}

// ---------------- WMMA fragment helpers (CDNA5 wave32 layouts) ----------------
// A fragment: 16x32 bf16 tile from LDS, row-major, leading dim ld.
// Lanes 0-15: M=lane, K in {0-7,16-23}; lanes 16-31: M=lane-16, K in {8-15,24-31}.
__device__ __forceinline__ v16bf load_a(const __bf16* t, int ld, int r0, int k0) {
  int lane = threadIdx.x & 31;
  int half = lane >> 4, m = lane & 15;
  const __bf16* p = t + (size_t)(r0 + m) * ld + k0;
  v16bf a;
#pragma unroll
  for (int c = 0; c < 16; ++c) {
    int k = (c & 7) + ((c >> 3) << 4) + 8 * half;
    a[c] = p[k];
  }
  return a;
}
// B fragment from pre-swizzled bf16 weights: fragment (kt, nt) is 32 lanes x 16 bf16
// contiguous; lane's 16 values are one 32-byte vector load.
__device__ __forceinline__ v16bf load_b_pre(const __bf16* Wp, int kt, int nt) {
  int lane = threadIdx.x & 31;
  return ((const v16bf*)Wp)[(size_t)(kt * 8 + nt) * 32 + lane];
}
__device__ __forceinline__ v8f wmma_bf16(v16bf a, v16bf b, v8f acc) {
  return __builtin_amdgcn_wmma_f32_16x16x32_bf16(false, a, false, b, (short)0, acc, false, false);
}

// ---------------- weight pre-swizzle: f32 KxN(=128) -> fragment-order bf16 ----------------
// out[((kt*8+nt)*32 + lane)*16 + c] = bf16( W[(kt*32 + (c&7) + ((c>>3)<<4) + 8*(lane>>4)) * 128 + nt*16 + (lane&15)] )
__global__ void pack_weights_kernel(const float* __restrict__ W, __bf16* __restrict__ out, int total) {
  int idx = blockIdx.x * blockDim.x + threadIdx.x;
  if (idx >= total) return;
  int c = idx & 15;
  int lane = (idx >> 4) & 31;
  int t = idx >> 9;
  int nt = t & 7;
  int kt = t >> 3;
  int k = kt * 32 + (c & 7) + ((c >> 3) << 4) + 8 * (lane >> 4);
  int n = nt * 16 + (lane & 15);
  out[idx] = f2bf(W[(size_t)k * HDIM + n]);
}

// ---------------- generic linear: Y = act(X @ W + b), N = 128 ----------------
// block = 256 threads (8 waves), 64 rows per block; wave = column tile, B held in regs.
template <int ACT, int K>
__global__ __launch_bounds__(256) void linear_kernel(
    const float* __restrict__ X, const __bf16* __restrict__ Wp,
    const float* __restrict__ bias, float* __restrict__ Y) {
  __shared__ __bf16 sx[64 * K];
  int tid = threadIdx.x;
  int rb = blockIdx.x * 64;
  for (int idx = tid; idx < 64 * K; idx += 256)
    sx[idx] = f2bf(X[(size_t)(rb + idx / K) * K + (idx % K)]);
  __syncthreads();
  int wave = tid >> 5, lane = tid & 31;
  int half = lane >> 4, nl = lane & 15;
  int ct = wave;
  int ng = ct * 16 + nl;
  v16bf b[K / 32];
#pragma unroll
  for (int k = 0; k < K / 32; ++k) b[k] = load_b_pre(Wp, k, ct);
  float bs = bias[ng];
  for (int rt = 0; rt < 4; ++rt) {
    v8f acc = {};
#pragma unroll
    for (int k = 0; k < K / 32; ++k)
      acc = wmma_bf16(load_a(sx, K, rt * 16, k * 32), b[k], acc);
#pragma unroll
    for (int i = 0; i < 8; ++i) {
      int m = rt * 16 + i + 8 * half;
      float v = acc[i] + bs;
      if (ACT) v = silu_f(v);
      Y[(size_t)(rb + m) * HDIM + ng] = v;
    }
  }
}

// ---------------- fused edge kernel ----------------
// 64 edges per block. Gather h[row]/h[col] -> e1 -> e2 -> c1 -> c2 -> atomic scatter.
__global__ __launch_bounds__(256) void edge_kernel(
    const int* __restrict__ row, const int* __restrict__ col,
    const float* __restrict__ h, const float* __restrict__ pos,
    const float* __restrict__ eattr,
    const __bf16* __restrict__ WaP, const __bf16* __restrict__ WbP,
    const float* __restrict__ wr, const float* __restrict__ We,
    const float* __restrict__ e1b,
    const __bf16* __restrict__ e2P, const float* __restrict__ e2b,
    const __bf16* __restrict__ c1P, const float* __restrict__ c1b,
    const float* __restrict__ c2v,
    float* __restrict__ agg, float* __restrict__ num, float* __restrict__ cnt) {
  __shared__ __bf16 s_hr[64 * HDIM];
  __shared__ __bf16 s_hc[64 * HDIM];
  __shared__ __bf16 s_m1[64 * HDIM];
  __shared__ __bf16 s_m[64 * HDIM];
  __shared__ __bf16 s_c[64 * HDIM];
  __shared__ float s_diff[64][3];
  __shared__ float s_rad[64];
  __shared__ float s_ea[64][4];
  __shared__ float s_w[64];

  int tid = threadIdx.x;
  int eb = blockIdx.x * 64;

  for (int idx = tid; idx < 64 * HDIM; idx += 256) {
    int e = idx >> 7, f = idx & (HDIM - 1);
    int r = row[eb + e], c = col[eb + e];
    s_hr[idx] = f2bf(h[(size_t)r * HDIM + f]);
    s_hc[idx] = f2bf(h[(size_t)c * HDIM + f]);
  }
  if (tid < 64) {
    int e = tid;
    int r = row[eb + e], c = col[eb + e];
    float rad = 0.0f;
#pragma unroll
    for (int d = 0; d < 3; ++d) {
      float dd = pos[(size_t)r * 3 + d] - pos[(size_t)c * 3 + d];
      s_diff[e][d] = dd;
      rad += dd * dd;
    }
    s_rad[e] = rad;
    if (eattr) {
#pragma unroll
      for (int j = 0; j < 4; ++j) s_ea[e][j] = eattr[(size_t)(eb + e) * 4 + j];
    }
  }
  __syncthreads();

  int wave = tid >> 5, lane = tid & 31;
  int half = lane >> 4, nl = lane & 15;
  int ct = wave;
  int ng = ct * 16 + nl;

  // GEMM1: m1 = silu(hr@Wa + hc@Wb + rad*wr + ea@We + b1)
  {
    v16bf bA[4], bB[4];
#pragma unroll
    for (int k = 0; k < 4; ++k) {
      bA[k] = load_b_pre(WaP, k, ct);
      bB[k] = load_b_pre(WbP, k, ct);
    }
    float bs = e1b[ng], wrn = wr[ng];
    float wen[4] = {0.f, 0.f, 0.f, 0.f};
    if (eattr) {
#pragma unroll
      for (int j = 0; j < 4; ++j) wen[j] = We[j * HDIM + ng];
    }
    for (int rt = 0; rt < 4; ++rt) {
      v8f acc = {};
#pragma unroll
      for (int k = 0; k < 4; ++k) {
        acc = wmma_bf16(load_a(s_hr, HDIM, rt * 16, k * 32), bA[k], acc);
        acc = wmma_bf16(load_a(s_hc, HDIM, rt * 16, k * 32), bB[k], acc);
      }
#pragma unroll
      for (int i = 0; i < 8; ++i) {
        int e = rt * 16 + i + 8 * half;
        float v = acc[i] + bs + s_rad[e] * wrn;
        if (eattr) {
#pragma unroll
          for (int j = 0; j < 4; ++j) v += s_ea[e][j] * wen[j];
        }
        s_m1[e * HDIM + ng] = f2bf(silu_f(v));
      }
    }
  }
  __syncthreads();

  // GEMM2: m = silu(m1 @ e2W + b2)
  {
    v16bf b[4];
#pragma unroll
    for (int k = 0; k < 4; ++k) b[k] = load_b_pre(e2P, k, ct);
    float bs = e2b[ng];
    for (int rt = 0; rt < 4; ++rt) {
      v8f acc = {};
#pragma unroll
      for (int k = 0; k < 4; ++k)
        acc = wmma_bf16(load_a(s_m1, HDIM, rt * 16, k * 32), b[k], acc);
#pragma unroll
      for (int i = 0; i < 8; ++i) {
        int e = rt * 16 + i + 8 * half;
        s_m[e * HDIM + ng] = f2bf(silu_f(acc[i] + bs));
      }
    }
  }
  __syncthreads();

  // GEMM3: c = silu(m @ c1W + c1b)
  {
    v16bf b[4];
#pragma unroll
    for (int k = 0; k < 4; ++k) b[k] = load_b_pre(c1P, k, ct);
    float bs = c1b[ng];
    for (int rt = 0; rt < 4; ++rt) {
      v8f acc = {};
#pragma unroll
      for (int k = 0; k < 4; ++k)
        acc = wmma_bf16(load_a(s_m, HDIM, rt * 16, k * 32), b[k], acc);
#pragma unroll
      for (int i = 0; i < 8; ++i) {
        int e = rt * 16 + i + 8 * half;
        s_c[e * HDIM + ng] = f2bf(silu_f(acc[i] + bs));
      }
    }
  }
  __syncthreads();

  // c2 dot: per-edge scalar weight
  if (tid < 64) {
    float s = 0.0f;
    for (int f = 0; f < HDIM; ++f) s += bf2f(s_c[tid * HDIM + f]) * c2v[f];
    s_w[tid] = s;
  }
  __syncthreads();

  // scatter (segment_sum via f32 atomics, lands in L2)
  for (int idx = tid; idx < 64 * HDIM; idx += 256) {
    int e = idx >> 7, f = idx & (HDIM - 1);
    atomicAdd(&agg[(size_t)row[eb + e] * HDIM + f], bf2f(s_m[e * HDIM + f]));
  }
  if (tid < 64) {
    int e = tid, r = row[eb + e];
    float w = s_w[e];
#pragma unroll
    for (int d = 0; d < 3; ++d) atomicAdd(&num[(size_t)r * 3 + d], s_diff[e][d] * w);
    atomicAdd(&cnt[r], 1.0f);
  }
}

// ---------------- fused node kernel: pos update + n1/n2 + residual ----------------
__global__ __launch_bounds__(256) void node_kernel(
    float* __restrict__ h, const float* __restrict__ agg,
    const float* __restrict__ num, const float* __restrict__ cnt,
    float* __restrict__ pos,
    const __bf16* __restrict__ n1P, const float* __restrict__ n1b,
    const __bf16* __restrict__ n2P, const float* __restrict__ n2b) {
  __shared__ __bf16 s_in[64 * 256];
  __shared__ __bf16 s_t1[64 * HDIM];
  int tid = threadIdx.x;
  int nb = blockIdx.x * 64;

  if (tid < 192) {
    int nloc = tid / 3, d = tid % 3;
    float c = fmaxf(cnt[nb + nloc], 1.0f);
    pos[(size_t)(nb + nloc) * 3 + d] += num[(size_t)(nb + nloc) * 3 + d] / c;
  }
  for (int idx = tid; idx < 64 * HDIM; idx += 256) {
    int r = idx >> 7, f = idx & (HDIM - 1);
    s_in[r * 256 + f]        = f2bf(h[(size_t)(nb + r) * HDIM + f]);
    s_in[r * 256 + HDIM + f] = f2bf(agg[(size_t)(nb + r) * HDIM + f]);
  }
  __syncthreads();

  int wave = tid >> 5, lane = tid & 31;
  int half = lane >> 4, nl = lane & 15;
  int ct = wave;
  int ng = ct * 16 + nl;

  // t1 = silu([h, agg] @ n1W + n1b)   (K = 256)
  {
    v16bf b[8];
#pragma unroll
    for (int k = 0; k < 8; ++k) b[k] = load_b_pre(n1P, k, ct);
    float bs = n1b[ng];
    for (int rt = 0; rt < 4; ++rt) {
      v8f acc = {};
#pragma unroll
      for (int k = 0; k < 8; ++k)
        acc = wmma_bf16(load_a(s_in, 256, rt * 16, k * 32), b[k], acc);
#pragma unroll
      for (int i = 0; i < 8; ++i) {
        int r = rt * 16 + i + 8 * half;
        s_t1[r * HDIM + ng] = f2bf(silu_f(acc[i] + bs));
      }
    }
  }
  __syncthreads();

  // h += t1 @ n2W + n2b
  {
    v16bf b[4];
#pragma unroll
    for (int k = 0; k < 4; ++k) b[k] = load_b_pre(n2P, k, ct);
    float bs = n2b[ng];
    for (int rt = 0; rt < 4; ++rt) {
      v8f acc = {};
#pragma unroll
      for (int k = 0; k < 4; ++k)
        acc = wmma_bf16(load_a(s_t1, HDIM, rt * 16, k * 32), b[k], acc);
#pragma unroll
      for (int i = 0; i < 8; ++i) {
        int r = rt * 16 + i + 8 * half;
        size_t gi = (size_t)(nb + r) * HDIM + ng;
        h[gi] = h[gi] + acc[i] + bs;
      }
    }
  }
}

// ---------------- pool graph construction ----------------
__global__ void build_pool_kernel(const float* __restrict__ h_main, const float* __restrict__ pos_main,
                                  const float* __restrict__ pos_cg,
                                  float* __restrict__ hf, float* __restrict__ pf,
                                  int atom_base, int n_at, int n_cg) {
  int i = blockIdx.x;
  int S = n_at + n_cg;
  int g = i / S, loc = i % S;
  int f = threadIdx.x;  // 128
  if (loc < n_at) {
    int src = g * NCPLX + atom_base + loc;
    hf[(size_t)i * HDIM + f] = h_main[(size_t)src * HDIM + f];
    if (f < 3) pf[(size_t)i * 3 + f] = pos_main[(size_t)src * 3 + f];
  } else {
    hf[(size_t)i * HDIM + f] = 0.0f;
    if (f < 3) pf[(size_t)i * 3 + f] = pos_cg[(size_t)(g * n_cg + (loc - n_at)) * 3 + f];
  }
}

__global__ void build_edges_kernel(const int* __restrict__ batch, int* __restrict__ rowO,
                                   int* __restrict__ colO, int n_at, int n_cg, int total) {
  int e = blockIdx.x * blockDim.x + threadIdx.x;
  if (e >= total) return;
  int g = e / n_at, j = e % n_at;
  int S = n_at + n_cg;
  rowO[e] = g * S + batch[e];
  colO[e] = g * S + j;
}

__global__ void gather_cg_kernel(const float* __restrict__ hf, float* __restrict__ out,
                                 int n_at, int n_cg) {
  int i = blockIdx.x;  // over G*n_cg
  int g = i / n_cg, c = i % n_cg;
  int f = threadIdx.x;  // 128
  out[(size_t)i * HDIM + f] = hf[(size_t)(g * (n_at + n_cg) + n_at + c) * HDIM + f];
}

// ---------------- host-side orchestration ----------------
// Param leaf order (jax pytree, sorted dict keys):
// top: egnn(37), lig_pool(37), lin_node(2), pro_pool(37)
// egnn: layers[3] x {c1W,c1b,c2,e1W,e1b,e2W,e2b,n1W,n1b,n2W,n2b}(11), winW, winb, woutW, woutb
struct EgnnP {
  const float *e1W[3], *e1b[3], *e2b[3];
  const float *c1b[3], *c2[3];
  const float *n1b[3], *n2b[3];
  const float *winb, *woutb;
  // packed bf16 GEMM weights
  const __bf16 *WaP[3], *WbP[3], *e2P[3], *c1P[3], *n1P[3], *n2P[3], *winP, *woutP;
};

static const __bf16* pack_w(const float* W, int K, __bf16*& cur, hipStream_t s) {
  int total = K * HDIM;
  pack_weights_kernel<<<(total + 255) / 256, 256, 0, s>>>(W, cur, total);
  const __bf16* r = cur;
  cur += total;
  return r;
}

static EgnnP get_egnn(void* const* d_in, int base, __bf16*& cur, hipStream_t s) {
  EgnnP p;
  for (int l = 0; l < 3; ++l) {
    int lb = base + l * 11;
    const float* c1W = (const float*)d_in[lb + 0];
    p.c1b[l] = (const float*)d_in[lb + 1];
    p.c2[l]  = (const float*)d_in[lb + 2];
    p.e1W[l] = (const float*)d_in[lb + 3];
    p.e1b[l] = (const float*)d_in[lb + 4];
    const float* e2W = (const float*)d_in[lb + 5];
    p.e2b[l] = (const float*)d_in[lb + 6];
    const float* n1W = (const float*)d_in[lb + 7];
    p.n1b[l] = (const float*)d_in[lb + 8];
    const float* n2W = (const float*)d_in[lb + 9];
    p.n2b[l] = (const float*)d_in[lb + 10];
    p.WaP[l] = pack_w(p.e1W[l], 128, cur, s);               // rows 0..127
    p.WbP[l] = pack_w(p.e1W[l] + 128 * HDIM, 128, cur, s);  // rows 128..255
    p.e2P[l] = pack_w(e2W, 128, cur, s);
    p.c1P[l] = pack_w(c1W, 128, cur, s);
    p.n1P[l] = pack_w(n1W, 256, cur, s);
    p.n2P[l] = pack_w(n2W, 128, cur, s);
  }
  p.winP  = pack_w((const float*)d_in[base + 33], 128, cur, s);
  p.winb  = (const float*)d_in[base + 34];
  p.woutP = pack_w((const float*)d_in[base + 35], 128, cur, s);
  p.woutb = (const float*)d_in[base + 36];
  return p;
}

static void run_egnn(const EgnnP& p, float* h, float* pos, const int* row, const int* col,
                     const float* eattr, int nNodes, int nE,
                     float* agg, float* num, float* cnt, hipStream_t stream) {
  linear_kernel<0, 128><<<nNodes / 64, 256, 0, stream>>>(h, p.winP, p.winb, h);
  for (int l = 0; l < 3; ++l) {
    hipMemsetAsync(agg, 0, (size_t)nNodes * HDIM * sizeof(float), stream);
    hipMemsetAsync(num, 0, (size_t)nNodes * 3 * sizeof(float), stream);
    hipMemsetAsync(cnt, 0, (size_t)nNodes * sizeof(float), stream);
    const float* wr = p.e1W[l] + 256 * HDIM;
    const float* We = p.e1W[l] + 257 * HDIM;  // only read when eattr != null
    edge_kernel<<<nE / 64, 256, 0, stream>>>(row, col, h, pos, eattr,
        p.WaP[l], p.WbP[l], wr, We, p.e1b[l],
        p.e2P[l], p.e2b[l], p.c1P[l], p.c1b[l], p.c2[l],
        agg, num, cnt);
    node_kernel<<<nNodes / 64, 256, 0, stream>>>(h, agg, num, cnt, pos,
        p.n1P[l], p.n1b[l], p.n2P[l], p.n2b[l]);
  }
  linear_kernel<0, 128><<<nNodes / 64, 256, 0, stream>>>(h, p.woutP, p.woutb, h);
}

extern "C" void kernel_launch(void* const* d_in, const int* in_sizes, int n_in,
                              void* d_out, int out_size, void* d_ws, size_t ws_size,
                              hipStream_t stream) {
  (void)in_sizes; (void)n_in; (void)out_size; (void)ws_size;

  const int EGNNB = 0, LIGB = 37, LINB = 74, PROB = 76;
  const float* x          = (const float*)d_in[113];
  const float* pos_in     = (const float*)d_in[114];
  const int*   eidx       = (const int*)d_in[115];
  const float* eattr      = (const float*)d_in[116];
  const int*   batch_lig  = (const int*)d_in[117];
  const int*   batch_pro  = (const int*)d_in[118];
  const float* pos_lig_cg = (const float*)d_in[119];
  const float* pos_pro_cg = (const float*)d_in[120];
  const float* linW = (const float*)d_in[LINB];
  const float* linb = (const float*)d_in[LINB + 1];

  // workspace carve-out (floats first, then bf16 packed-weight arena)
  const int NL = G_ * 144;    // 2304 lig pool nodes
  const int NP = G_ * 1984;   // 31744 pro pool nodes
  const int EL = G_ * 128;    // 2048 lig pool edges
  const int EP = G_ * 1920;   // 30720 pro pool edges
  float* w = (float*)d_ws;
  size_t o = 0;
  float* pos_main = w + o; o += (size_t)NNODES * 3;
  float* h_main   = w + o; o += (size_t)NNODES * HDIM;
  float* agg      = w + o; o += (size_t)NNODES * HDIM;  // shared across EGNNs
  float* num      = w + o; o += (size_t)NNODES * 3;
  float* cnt      = w + o; o += (size_t)NNODES;
  float* h_lig    = w + o; o += (size_t)NL * HDIM;
  float* p_lig    = w + o; o += (size_t)NL * 3;
  float* h_pro    = w + o; o += (size_t)NP * HDIM;
  float* p_pro    = w + o; o += (size_t)NP * 3;
  int* rowL = (int*)(w + o); o += EL;
  int* colL = (int*)(w + o); o += EL;
  int* rowP = (int*)(w + o); o += EP;
  int* colP = (int*)(w + o); o += EP;
  o = (o + 7) & ~(size_t)7;  // 32B-align the bf16 arena
  __bf16* cur = (__bf16*)(w + o);

  // pre-swizzle all GEMM weights to fragment-order bf16 (tiny: ~2.3 MB total)
  const __bf16* linP = pack_w(linW, 64, cur, stream);
  EgnnP pm = get_egnn(d_in, EGNNB, cur, stream);
  EgnnP pl = get_egnn(d_in, LIGB, cur, stream);
  EgnnP pp = get_egnn(d_in, PROB, cur, stream);

  // --- main EGNN over the complex graph ---
  hipMemcpyAsync(pos_main, pos_in, (size_t)NNODES * 3 * sizeof(float),
                 hipMemcpyDeviceToDevice, stream);
  linear_kernel<1, 64><<<NNODES / 64, 256, 0, stream>>>(x, linP, linb, h_main);
  run_egnn(pm, h_main, pos_main, eidx, eidx + NEDGE, eattr, NNODES, NEDGE,
           agg, num, cnt, stream);

  // --- ligand pool EGNN ---
  build_pool_kernel<<<NL, 128, 0, stream>>>(h_main, pos_main, pos_lig_cg,
                                            h_lig, p_lig, 0, 128, 16);
  build_edges_kernel<<<(EL + 255) / 256, 256, 0, stream>>>(batch_lig, rowL, colL, 128, 16, EL);
  run_egnn(pl, h_lig, p_lig, rowL, colL, nullptr, NL, EL, agg, num, cnt, stream);

  // --- protein pool EGNN ---
  build_pool_kernel<<<NP, 128, 0, stream>>>(h_main, pos_main, pos_pro_cg,
                                            h_pro, p_pro, 128, 1920, 64);
  build_edges_kernel<<<(EP + 255) / 256, 256, 0, stream>>>(batch_pro, rowP, colP, 1920, 64, EP);
  run_egnn(pp, h_pro, p_pro, rowP, colP, nullptr, NP, EP, agg, num, cnt, stream);

  // --- outputs: (x_lig_cg, x_pro_cg, pos_lig_cg, pos_pro_cg) flat ---
  float* out = (float*)d_out;
  gather_cg_kernel<<<G_ * 16, 128, 0, stream>>>(h_lig, out, 128, 16);
  gather_cg_kernel<<<G_ * 64, 128, 0, stream>>>(h_pro, out + (size_t)G_ * 16 * HDIM, 1920, 64);
  size_t posoff = (size_t)G_ * 16 * HDIM + (size_t)G_ * 64 * HDIM;  // 163840
  hipMemcpyAsync(out + posoff, pos_lig_cg, (size_t)G_ * 16 * 3 * sizeof(float),
                 hipMemcpyDeviceToDevice, stream);
  hipMemcpyAsync(out + posoff + G_ * 16 * 3, pos_pro_cg, (size_t)G_ * 64 * 3 * sizeof(float),
                 hipMemcpyDeviceToDevice, stream);
}
